// BackBoneV2_54494545051943
// MI455X (gfx1250) — matrix operand order repaired and checked by reading, not compiled, not run on hardware
//
#include <hip/hip_runtime.h>

// ---------------------------------------------------------------------------
// Types for CDNA5 WMMA (wave32)
// ---------------------------------------------------------------------------
typedef __bf16 bf16_t;
typedef bf16_t v16bf __attribute__((ext_vector_type(16)));
typedef bf16_t v8bf  __attribute__((ext_vector_type(8)));
typedef float  v8f   __attribute__((ext_vector_type(8)));

#define DD   64
#define VD_  4525
#define VL_  753
#define ND_  (DD * VD_)      // 289600
#define NLN_ (DD * VL_)      // 48192
#define HH   128
#define EMB_ 10
#define TV_  100
#define ED_  32768
#define ET_  131072
#define KC_  256
#define LENC 6
#define FF_  2048

// ---------------------------------------------------------------------------
// Pack a row-major f32 weight [Ksrc x N] into bf16 WMMA B-fragment order,
// zero-padding K up to Kpad (multiple of 32). Layout per 16x16x32 tile
// (kt, nt): lane l holds 16 contiguous bf16 = B[kt*32 + (l>=16?16:0)+j][nt*16+(l&15)].
// ---------------------------------------------------------------------------
__global__ void pack_b_kernel(const float* __restrict__ B, bf16_t* __restrict__ Bp,
                              int Ksrc, int Kpad, int N) {
  int t = blockIdx.x * blockDim.x + threadIdx.x;
  int total = (Kpad >> 5) * (N >> 4) * 32;
  if (t >= total) return;
  int l = t & 31;
  int tile = t >> 5;
  int ntile = N >> 4;
  int nt = tile % ntile;
  int kt = tile / ntile;
  int n = nt * 16 + (l & 15);
  int kbase = kt * 32 + (l >> 4) * 16;
  bf16_t* dst = Bp + (size_t)t * 16;
#pragma unroll
  for (int j = 0; j < 16; ++j) {
    int k = kbase + j;
    float v = (k < Ksrc) ? B[(size_t)k * N + n] : 0.0f;
    dst[j] = (bf16_t)v;
  }
}

// ---------------------------------------------------------------------------
// bf16 GEMM, 16x(16*NT) output strip per wave. NT=8 => one A fragment feeds
// 8 v_wmma per K-step: A HBM traffic is exactly 1x for N=128 (packed B is
// L2-resident), and load:wmma ratio is 2.25 b128 loads per v_wmma.
// Fused bias / residual-add / relu / optional f32 store / optional bf16 store.
// Requires: Kp % 32 == 0, N % (16*NT) == 0. grid = (ceil(M/16), N/(16*NT)).
// ---------------------------------------------------------------------------
template <int NT>
__global__ void gemm_wmma_bf16(const bf16_t* __restrict__ A,
                               const bf16_t* __restrict__ Bp,
                               const float* __restrict__ bias,
                               const float* __restrict__ addsrc,
                               float* __restrict__ C,
                               bf16_t* __restrict__ Cb,
                               int M, int N, int Kp, int relu) {
  const int tile_m  = blockIdx.x;
  const int tile_n0 = blockIdx.y * NT;     // first 16-col tile of this strip
  const int l  = threadIdx.x;              // 0..31
  const int lm = l & 15;
  const int hf = l >> 4;                   // 0 or 1

  int mrow = tile_m * 16 + lm;
  if (mrow >= M) mrow = M - 1;             // clamp (stores guarded below)

  const int ktiles = Kp >> 5;
  const bf16_t* arow = A + (size_t)mrow * Kp;
  const size_t  bstride = (size_t)(N >> 4) * 32 * 16;   // elements per k-tile step
  const bf16_t* bpbase = Bp + ((size_t)tile_n0 * 32 + l) * 16;
  const int k0 = hf * 16;

  v8f acc[NT];
#pragma unroll
  for (int c = 0; c < NT; ++c)
#pragma unroll
    for (int j = 0; j < 8; ++j) acc[c][j] = 0.f;

  for (int kt = 0; kt < ktiles; ++kt) {
    const bf16_t* ap = arow + kt * 32;
    v8bf alo = *(const v8bf*)(ap + k0);
    v8bf ahi = *(const v8bf*)(ap + 8 + k0);
    const bf16_t* bp = bpbase + (size_t)kt * bstride;
    if (kt + 1 < ktiles) {
      __builtin_prefetch(ap + 32, 0, 1);               // global_prefetch_b8
      __builtin_prefetch(bp + bstride, 0, 1);
    }
    v16bf af;
#pragma unroll
    for (int j = 0; j < 8; ++j) { af[j] = alo[j]; af[8 + j] = ahi[j]; }

#pragma unroll
    for (int c = 0; c < NT; ++c) {
      const bf16_t* bpc = bp + (size_t)c * 32 * 16;
      v8bf b0 = *(const v8bf*)bpc;
      v8bf b1 = *(const v8bf*)(bpc + 8);
      v16bf bfv;
#pragma unroll
      for (int j = 0; j < 8; ++j) { bfv[j] = b0[j]; bfv[8 + j] = b1[j]; }
      acc[c] = __builtin_amdgcn_wmma_f32_16x16x32_bf16(
          false, af, false, bfv, (short)0, acc[c], false, false);
    }
  }

#pragma unroll
  for (int c = 0; c < NT; ++c) {
    const int n = (tile_n0 + c) * 16 + lm;
#pragma unroll
    for (int j = 0; j < 8; ++j) {
      int m = tile_m * 16 + hf * 8 + j;
      if (m < M) {
        float v = acc[c][j];
        if (bias)   v += bias[n];
        if (addsrc) v += addsrc[(size_t)m * N + n];
        if (relu)   v = v > 0.0f ? v : 0.0f;
        if (C)  C[(size_t)m * N + n] = v;
        if (Cb) Cb[(size_t)m * N + n] = (bf16_t)v;
      }
    }
  }
}

// ---------------------------------------------------------------------------
// Node embedding: [id_emb?] ++ token embeddings ++ floats@float_w, zero-padded
// to kpad, written as bf16 row-major features.
// ---------------------------------------------------------------------------
__global__ void embed_nodes(const int* __restrict__ toks, int ntok,
                            const int* __restrict__ node_id,
                            const float* __restrict__ id_emb,
                            const float* __restrict__ floats, int nflt,
                            const float* __restrict__ tok_emb,
                            const float* __restrict__ fw,
                            bf16_t* __restrict__ out, int kpad, int n) {
  int i = blockIdx.x * blockDim.x + threadIdx.x;
  if (i >= n) return;
  bf16_t* o = out + (size_t)i * kpad;
  int p = 0;
  if (node_id) {
    const float* e = id_emb + (size_t)node_id[i] * EMB_;
#pragma unroll
    for (int j = 0; j < EMB_; ++j) o[p++] = (bf16_t)e[j];
  }
  for (int t = 0; t < ntok; ++t) {
    const float* e = tok_emb + ((size_t)t * TV_ + toks[(size_t)i * ntok + t]) * EMB_;
#pragma unroll
    for (int j = 0; j < EMB_; ++j) o[p++] = (bf16_t)e[j];
  }
  float fe[EMB_];
#pragma unroll
  for (int j = 0; j < EMB_; ++j) fe[j] = 0.0f;
  for (int f = 0; f < nflt; ++f) {
    float v = floats[(size_t)i * nflt + f];
#pragma unroll
    for (int j = 0; j < EMB_; ++j) fe[j] += v * fw[f * EMB_ + j];
  }
#pragma unroll
  for (int j = 0; j < EMB_; ++j) o[p++] = (bf16_t)fe[j];
  for (; p < kpad; ++p) o[p] = (bf16_t)0.0f;
}

// out[e,h] = x[idx[e],h] + eemb[e,h]; 8 bf16 (16B) per thread, H=128 -> 16 chunks.
__global__ void gather_add_bf16(const bf16_t* __restrict__ x,
                                const int* __restrict__ idx,
                                const bf16_t* __restrict__ eemb,
                                bf16_t* __restrict__ out, int nchunks) {
  int t = blockIdx.x * blockDim.x + threadIdx.x;
  if (t >= nchunks) return;
  int e = t >> 4, c = t & 15;
  const v8bf xv = *(const v8bf*)(x + (size_t)idx[e] * HH + c * 8);
  const v8bf ev = *(const v8bf*)(eemb + (size_t)t * 8);
  v8bf ov;
#pragma unroll
  for (int j = 0; j < 8; ++j) ov[j] = (bf16_t)((float)xv[j] + (float)ev[j]);
  *(v8bf*)(out + (size_t)t * 8) = ov;
}

__global__ void count_kernel(const int* __restrict__ idx, int* __restrict__ cnt, int E) {
  int e = blockIdx.x * blockDim.x + threadIdx.x;
  if (e < E) atomicAdd(&cnt[idx[e]], 1);
}

// 4 floats per thread (b128 source load + 4 atomics)
__global__ void scatter_add(const float* __restrict__ src, const int* __restrict__ idx,
                            float* __restrict__ acc, int nquads) {
  int t = blockIdx.x * blockDim.x + threadIdx.x;
  if (t >= nquads) return;
  int e = t >> 5, h = (t & 31) * 4;
  const float4 v = *(const float4*)(src + (size_t)t * 4);
  float* a = acc + (size_t)idx[e] * HH + h;
  atomicAdd(a + 0, v.x);
  atomicAdd(a + 1, v.y);
  atomicAdd(a + 2, v.z);
  atomicAdd(a + 3, v.w);
}

__global__ void div_cnt(float* __restrict__ acc, const int* __restrict__ cnt, int nquads) {
  int t = blockIdx.x * blockDim.x + threadIdx.x;
  if (t >= nquads) return;
  int c = cnt[t >> 5];
  float inv = 1.0f / (float)(c > 1 ? c : 1);
  float4 v = *(float4*)(acc + (size_t)t * 4);
  v.x *= inv; v.y *= inv; v.z *= inv; v.w *= inv;
  *(float4*)(acc + (size_t)t * 4) = v;
}

__global__ void add_div(float* __restrict__ dst, const float* __restrict__ src,
                        const int* __restrict__ cnt, int nquads) {
  int t = blockIdx.x * blockDim.x + threadIdx.x;
  if (t >= nquads) return;
  int c = cnt[t >> 5];
  float inv = 1.0f / (float)(c > 1 ? c : 1);
  const float4 s = *(const float4*)(src + (size_t)t * 4);
  float4 d = *(float4*)(dst + (size_t)t * 4);
  d.x += s.x * inv; d.y += s.y * inv; d.z += s.z * inv; d.w += s.w * inv;
  *(float4*)(dst + (size_t)t * 4) = d;
}

// ---------------------------------------------------------------------------
// Causal multi-head attention for D=64 tokens, NH=8 heads, HD=16.
// grid = 8 heads, block = 64 threads (one per query row).
// ---------------------------------------------------------------------------
__global__ void attention_kernel(const float* __restrict__ qkv,   // [64,384]
                                 float* __restrict__ o_f32,       // [64,128]
                                 bf16_t* __restrict__ o_bf) {
  int hd = blockIdx.x;
  int qi = threadIdx.x;
  __shared__ float Qs[64][16], Ks[64][16], Vs[64][16];
#pragma unroll
  for (int j = 0; j < 16; ++j) {
    Qs[qi][j] = qkv[qi * 384 + hd * 16 + j];
    Ks[qi][j] = qkv[qi * 384 + 128 + hd * 16 + j];
    Vs[qi][j] = qkv[qi * 384 + 256 + hd * 16 + j];
  }
  __syncthreads();
  __shared__ float sc[64][64];
  float mx = -1e30f;
  for (int k = 0; k <= qi; ++k) {
    float s = 0.0f;
#pragma unroll
    for (int j = 0; j < 16; ++j) s += Qs[qi][j] * Ks[k][j];
    s *= 0.25f;
    sc[qi][k] = s;
    mx = fmaxf(mx, s);
  }
  float den = 0.0f;
  for (int k = 0; k <= qi; ++k) {
    float p = __expf(sc[qi][k] - mx);
    sc[qi][k] = p;
    den += p;
  }
  float inv = 1.0f / den;
#pragma unroll
  for (int j = 0; j < 16; ++j) {
    float acc = 0.0f;
    for (int k = 0; k <= qi; ++k) acc += sc[qi][k] * Vs[k][j];
    acc *= inv;
    o_f32[qi * 128 + hd * 16 + j] = acc;
    o_bf[qi * 128 + hd * 16 + j] = (bf16_t)acc;
  }
}

// LayerNorm over H=128; grid = rows, block = 128.
__global__ void layernorm_kernel(const float* __restrict__ x, const float* __restrict__ sb,
                                 float* __restrict__ out_f32, bf16_t* __restrict__ out_bf) {
  int r = blockIdx.x, t = threadIdx.x;
  __shared__ float red[128];
  float v = x[r * 128 + t];
  red[t] = v;
  __syncthreads();
  for (int s = 64; s > 0; s >>= 1) { if (t < s) red[t] += red[t + s]; __syncthreads(); }
  float mu = red[0] * (1.0f / 128.0f);
  __syncthreads();
  float d = v - mu;
  red[t] = d * d;
  __syncthreads();
  for (int s = 64; s > 0; s >>= 1) { if (t < s) red[t] += red[t + s]; __syncthreads(); }
  float var = red[0] * (1.0f / 128.0f);
  float y = sb[t] * d * rsqrtf(var + 1e-5f) + sb[128 + t];
  out_f32[r * 128 + t] = y;
  out_bf[r * 128 + t] = (bf16_t)y;
}

__global__ void hlp_kernel(const float* __restrict__ h, const float* __restrict__ lp,
                           float* __restrict__ hlp) {
  int t = blockIdx.x * blockDim.x + threadIdx.x;
  if (t >= DD * HH) return;
  hlp[t] = h[t] * lp[t & 127];
}

// logits[d,k] = sum_h x_d_bf16[d*VD + cand[d,k], h] * hlp[d,h]   (d in 0..62)
// Reads the bf16 state copy: no f32 x_d buffer needs to exist at all.
__global__ void logits_kernel(const bf16_t* __restrict__ xd, const float* __restrict__ hlp,
                              const int* __restrict__ cand, float* __restrict__ out) {
  int t = blockIdx.x * blockDim.x + threadIdx.x;
  if (t >= (DD - 1) * KC_) return;
  int d = t / KC_;
  const bf16_t* item = xd + ((size_t)d * VD_ + cand[t]) * HH;
  const float* hv = hlp + d * HH;
  float s = 0.0f;
#pragma unroll
  for (int c = 0; c < 16; ++c) {
    const v8bf iv = *(const v8bf*)(item + c * 8);
#pragma unroll
    for (int j = 0; j < 8; ++j) s += (float)iv[j] * hv[c * 8 + j];
  }
  out[t] = s;
}

// ---------------------------------------------------------------------------
extern "C" void kernel_launch(void* const* d_in, const int* in_sizes, int n_in,
                              void* d_out, int out_size, void* d_ws, size_t ws_size,
                              hipStream_t stream) {
  (void)in_sizes; (void)n_in; (void)out_size; (void)ws_size;

  const int*   drug_tokens  = (const int*)d_in[0];
  const int*   drug_node_id = (const int*)d_in[1];
  const float* drug_floats  = (const float*)d_in[2];
  const int*   lab_tokens   = (const int*)d_in[3];
  const int*   lab_node_id  = (const int*)d_in[4];
  const float* lab_floats   = (const float*)d_in[5];
  const int*   adm_tokens   = (const int*)d_in[6];
  const float* adm_floats   = (const float*)d_in[7];
  const int*   did_tokens   = (const int*)d_in[8];
  const float* did_floats   = (const float*)d_in[9];
  const int*   did_src      = (const int*)d_in[10];
  const int*   did_dst      = (const int*)d_in[11];
  const int*   took_tokens  = (const int*)d_in[12];
  const float* took_floats  = (const float*)d_in[13];
  const int*   took_src     = (const int*)d_in[14];
  const int*   took_dst     = (const int*)d_in[15];
  const int*   cand_idx     = (const int*)d_in[16];
  const float* drug_id_emb  = (const float*)d_in[17];
  const float* lab_id_emb   = (const float*)d_in[18];
  const float* drug_tok_emb = (const float*)d_in[19];
  const float* lab_tok_emb  = (const float*)d_in[20];
  const float* adm_tok_emb  = (const float*)d_in[21];
  const float* did_tok_emb  = (const float*)d_in[22];
  const float* took_tok_emb = (const float*)d_in[23];
  const float* drug_float_w = (const float*)d_in[24];
  const float* lab_float_w  = (const float*)d_in[25];
  const float* adm_float_w  = (const float*)d_in[26];
  const float* did_float_w  = (const float*)d_in[27];
  const float* took_float_w = (const float*)d_in[28];
  const float* drug_align   = (const float*)d_in[29];
  const float* lab_align    = (const float*)d_in[30];
  const float* adm_align    = (const float*)d_in[31];
  const float* did_align    = (const float*)d_in[32];
  const float* took_align   = (const float*)d_in[33];
  const float* gnn_w_msg    = (const float*)d_in[34];
  const float* gnn_w_self   = (const float*)d_in[35];
  const float* tr_qkv_w     = (const float*)d_in[36];
  const float* tr_qkv_b     = (const float*)d_in[37];
  const float* tr_out_w     = (const float*)d_in[38];
  const float* tr_out_b     = (const float*)d_in[39];
  const float* tr_ln1       = (const float*)d_in[40];
  const float* tr_ff1_w     = (const float*)d_in[41];
  const float* tr_ff1_b     = (const float*)d_in[42];
  const float* tr_ff2_w     = (const float*)d_in[43];
  const float* tr_ff2_b     = (const float*)d_in[44];
  const float* tr_ln2       = (const float*)d_in[45];
  const float* lp_w         = (const float*)d_in[46];
  float* out = (float*)d_out;

  // --- bump allocator over workspace ---
  char* wp = (char*)d_ws;
  size_t off = 0;
  auto take = [&](size_t bytes) -> void* {
    void* r = wp + off;
    off = (off + bytes + 255) & ~(size_t)255;
    return r;
  };

  // packed weights (bf16)
  bf16_t* pDrugAl = (bf16_t*)take(64 * 128 * 2);
  bf16_t* pLabAl  = (bf16_t*)take(64 * 128 * 2);
  bf16_t* pAdmAl  = (bf16_t*)take(64 * 128 * 2);
  bf16_t* pDidAl  = (bf16_t*)take(32 * 128 * 2);
  bf16_t* pTookAl = (bf16_t*)take(32 * 128 * 2);
  bf16_t* pMsg    = (bf16_t*)take((size_t)8 * 128 * 128 * 2);
  bf16_t* pSelf   = (bf16_t*)take((size_t)6 * 128 * 128 * 2);
  bf16_t* pQkv    = (bf16_t*)take((size_t)LENC * 128 * 384 * 2);
  bf16_t* pOutW   = (bf16_t*)take((size_t)LENC * 128 * 128 * 2);
  bf16_t* pFf1    = (bf16_t*)take((size_t)LENC * 128 * FF_ * 2);
  bf16_t* pFf2    = (bf16_t*)take((size_t)LENC * FF_ * 128 * 2);

  // features / states (all large state kept in bf16 only)
  bf16_t* featD   = (bf16_t*)take((size_t)ND_ * 64 * 2);
  bf16_t* featL   = (bf16_t*)take((size_t)NLN_ * 64 * 2);
  bf16_t* featA   = (bf16_t*)take((size_t)DD * 64 * 2);
  bf16_t* featDid = (bf16_t*)take((size_t)ED_ * 32 * 2);
  bf16_t* featTk  = (bf16_t*)take((size_t)ET_ * 32 * 2);
  bf16_t* xd_bf[2] = { (bf16_t*)take((size_t)ND_ * HH * 2),
                       (bf16_t*)take((size_t)ND_ * HH * 2) };
  bf16_t* xl_bf[2] = { (bf16_t*)take((size_t)NLN_ * HH * 2),
                       (bf16_t*)take((size_t)NLN_ * HH * 2) };
  float*  xa_f32  = (float*)take((size_t)DD * HH * 4);
  bf16_t* xa_bf[2] = { (bf16_t*)take((size_t)DD * HH * 2),
                       (bf16_t*)take((size_t)DD * HH * 2) };
  bf16_t* ed_bf   = (bf16_t*)take((size_t)ED_ * HH * 2);
  bf16_t* et_bf   = (bf16_t*)take((size_t)ET_ * HH * 2);
  bf16_t* ga      = (bf16_t*)take((size_t)ET_ * HH * 2);   // gather scratch (max E)
  float*  tmp     = (float*)take((size_t)ET_ * HH * 4);    // msg GEMM output
  float*  m_d     = (float*)take((size_t)ND_ * HH * 4);
  float*  m_l     = (float*)take((size_t)NLN_ * HH * 4);
  float*  m_a     = (float*)take((size_t)DD * HH * 4);
  float*  m_part  = (float*)take((size_t)DD * HH * 4);
  int* cnt_did_dst  = (int*)take((size_t)NLN_ * 4);
  int* cnt_took_dst = (int*)take((size_t)ND_ * 4);
  int* cnt_did_src  = (int*)take(DD * 4);
  int* cnt_took_src = (int*)take(DD * 4);
  float*  qkv_f32 = (float*)take((size_t)DD * 384 * 4);
  float*  o_f32   = (float*)take((size_t)DD * HH * 4);
  bf16_t* o_bf    = (bf16_t*)take((size_t)DD * HH * 2);
  float*  preln   = (float*)take((size_t)DD * HH * 4);
  bf16_t* f_bf    = (bf16_t*)take((size_t)DD * FF_ * 2);
  float*  hlp     = (float*)take((size_t)DD * HH * 4);

  auto pack = [&](const float* src, bf16_t* dst, int Ksrc, int Kpad, int N) {
    int total = (Kpad >> 5) * (N >> 4) * 32;
    pack_b_kernel<<<(total + 255) / 256, 256, 0, stream>>>(src, dst, Ksrc, Kpad, N);
  };
  // NT=8: 16x128 strip per wave; valid for all N used here (128, 384, 2048).
  auto gemm = [&](const bf16_t* A, const bf16_t* Bp, const float* bias, const float* add,
                  float* C, bf16_t* Cb, int M, int N, int Kp, int relu) {
    dim3 g((M + 15) / 16, N / 128);
    gemm_wmma_bf16<8><<<g, 32, 0, stream>>>(A, Bp, bias, add, C, Cb, M, N, Kp, relu);
  };
  auto grid1 = [](int n) { return dim3((n + 255) / 256); };

  // --- 1. pack all weights into WMMA B-fragment order ---
  pack(drug_align, pDrugAl, 50, 64, 128);
  pack(lab_align,  pLabAl,  40, 64, 128);
  pack(adm_align,  pAdmAl,  50, 64, 128);
  pack(did_align,  pDidAl,  30, 32, 128);
  pack(took_align, pTookAl, 30, 32, 128);
  for (int i = 0; i < 8; ++i)
    pack(gnn_w_msg + (size_t)i * 128 * 128, pMsg + (size_t)i * 128 * 128, 128, 128, 128);
  for (int i = 0; i < 6; ++i)
    pack(gnn_w_self + (size_t)i * 128 * 128, pSelf + (size_t)i * 128 * 128, 128, 128, 128);
  for (int i = 0; i < LENC; ++i) {
    pack(tr_qkv_w + (size_t)i * 128 * 384, pQkv + (size_t)i * 128 * 384, 128, 128, 384);
    pack(tr_out_w + (size_t)i * 128 * 128, pOutW + (size_t)i * 128 * 128, 128, 128, 128);
    pack(tr_ff1_w + (size_t)i * 128 * FF_, pFf1 + (size_t)i * 128 * FF_, 128, 128, FF_);
    pack(tr_ff2_w + (size_t)i * FF_ * 128, pFf2 + (size_t)i * FF_ * 128, FF_, FF_, 128);
  }

  // --- 2. node / edge feature embedding (bf16, K padded) ---
  embed_nodes<<<grid1(ND_), 256, 0, stream>>>(drug_tokens, 3, drug_node_id, drug_id_emb,
      drug_floats, 2, drug_tok_emb, drug_float_w, featD, 64, ND_);
  embed_nodes<<<grid1(NLN_), 256, 0, stream>>>(lab_tokens, 2, lab_node_id, lab_id_emb,
      lab_floats, 1, lab_tok_emb, lab_float_w, featL, 64, NLN_);
  embed_nodes<<<grid1(DD), 256, 0, stream>>>(adm_tokens, 4, nullptr, nullptr,
      adm_floats, 3, adm_tok_emb, adm_float_w, featA, 64, DD);
  embed_nodes<<<grid1(ED_), 256, 0, stream>>>(did_tokens, 2, nullptr, nullptr,
      did_floats, 2, did_tok_emb, did_float_w, featDid, 32, ED_);
  embed_nodes<<<grid1(ET_), 256, 0, stream>>>(took_tokens, 2, nullptr, nullptr,
      took_floats, 1, took_tok_emb, took_float_w, featTk, 32, ET_);

  // --- 3. alignment GEMMs (bf16-only outputs; no dead f32 traffic) ---
  gemm(featD, pDrugAl, nullptr, nullptr, nullptr, xd_bf[0], ND_, 128, 64, 0);
  gemm(featL, pLabAl,  nullptr, nullptr, nullptr, xl_bf[0], NLN_, 128, 64, 0);
  gemm(featA, pAdmAl,  nullptr, nullptr, xa_f32, xa_bf[0], DD, 128, 64, 0);
  gemm(featDid, pDidAl,  nullptr, nullptr, nullptr, ed_bf, ED_, 128, 32, 0);
  gemm(featTk,  pTookAl, nullptr, nullptr, nullptr, et_bf, ET_, 128, 32, 0);

  // --- 4. segment counts (fixed per graph, recomputed each call) ---
  hipMemsetAsync(cnt_did_dst, 0, (size_t)NLN_ * 4, stream);
  hipMemsetAsync(cnt_took_dst, 0, (size_t)ND_ * 4, stream);
  hipMemsetAsync(cnt_did_src, 0, DD * 4, stream);
  hipMemsetAsync(cnt_took_src, 0, DD * 4, stream);
  count_kernel<<<grid1(ED_), 256, 0, stream>>>(did_dst, cnt_did_dst, ED_);
  count_kernel<<<grid1(ET_), 256, 0, stream>>>(took_dst, cnt_took_dst, ET_);
  count_kernel<<<grid1(ED_), 256, 0, stream>>>(did_src, cnt_did_src, ED_);
  count_kernel<<<grid1(ET_), 256, 0, stream>>>(took_src, cnt_took_src, ET_);

  // --- 5. GNN layers ---
  int cur = 0;
  for (int l = 0; l < 2; ++l) {
    int nxt = 1 - cur;
    int relu = (l == 0) ? 1 : 0;
    bf16_t* wMsg = pMsg + (size_t)l * 4 * 128 * 128;
    bf16_t* wSelf = pSelf + (size_t)l * 3 * 128 * 128;

    // m_l = seg_mean((x_a[did_src]+e_did) @ W0, did_dst, NLN)
    hipMemsetAsync(m_l, 0, (size_t)NLN_ * HH * 4, stream);
    gather_add_bf16<<<grid1(ED_ * 16), 256, 0, stream>>>(xa_bf[cur], did_src, ed_bf, ga, ED_ * 16);
    gemm(ga, wMsg + 0 * 128 * 128, nullptr, nullptr, tmp, nullptr, ED_, 128, 128, 0);
    scatter_add<<<grid1(ED_ * 32), 256, 0, stream>>>(tmp, did_dst, m_l, ED_ * 32);
    div_cnt<<<grid1(NLN_ * 32), 256, 0, stream>>>(m_l, cnt_did_dst, NLN_ * 32);

    // m_d = seg_mean((x_a[took_src]+e_took) @ W1, took_dst, ND)
    hipMemsetAsync(m_d, 0, (size_t)ND_ * HH * 4, stream);
    gather_add_bf16<<<grid1(ET_ * 16), 256, 0, stream>>>(xa_bf[cur], took_src, et_bf, ga, ET_ * 16);
    gemm(ga, wMsg + 1 * 128 * 128, nullptr, nullptr, tmp, nullptr, ET_, 128, 128, 0);
    scatter_add<<<grid1(ET_ * 32), 256, 0, stream>>>(tmp, took_dst, m_d, ET_ * 32);
    div_cnt<<<grid1(ND_ * 32), 256, 0, stream>>>(m_d, cnt_took_dst, ND_ * 32);

    // m_a = seg_mean((x_l[did_dst]+e_did) @ W2, did_src, D)
    //     + seg_mean((x_d[took_dst]+e_took) @ W3, took_src, D)
    hipMemsetAsync(m_a, 0, (size_t)DD * HH * 4, stream);
    hipMemsetAsync(m_part, 0, (size_t)DD * HH * 4, stream);
    gather_add_bf16<<<grid1(ED_ * 16), 256, 0, stream>>>(xl_bf[cur], did_dst, ed_bf, ga, ED_ * 16);
    gemm(ga, wMsg + 2 * 128 * 128, nullptr, nullptr, tmp, nullptr, ED_, 128, 128, 0);
    scatter_add<<<grid1(ED_ * 32), 256, 0, stream>>>(tmp, did_src, m_part, ED_ * 32);
    add_div<<<grid1(DD * 32), 256, 0, stream>>>(m_a, m_part, cnt_did_src, DD * 32);
    hipMemsetAsync(m_part, 0, (size_t)DD * HH * 4, stream);
    gather_add_bf16<<<grid1(ET_ * 16), 256, 0, stream>>>(xd_bf[cur], took_dst, et_bf, ga, ET_ * 16);
    gemm(ga, wMsg + 3 * 128 * 128, nullptr, nullptr, tmp, nullptr, ET_, 128, 128, 0);
    scatter_add<<<grid1(ET_ * 32), 256, 0, stream>>>(tmp, took_src, m_part, ET_ * 32);
    add_div<<<grid1(DD * 32), 256, 0, stream>>>(m_a, m_part, cnt_took_src, DD * 32);

    // self-updates (fused residual msg add + relu + bf16 recast; bf16-only state)
    gemm(xa_bf[cur], wSelf + 0 * 128 * 128, nullptr, m_a, xa_f32, xa_bf[nxt], DD, 128, 128, relu);
    gemm(xd_bf[cur], wSelf + 1 * 128 * 128, nullptr, m_d, nullptr, xd_bf[nxt], ND_, 128, 128, relu);
    gemm(xl_bf[cur], wSelf + 2 * 128 * 128, nullptr, m_l, nullptr, xl_bf[nxt], NLN_, 128, 128, relu);
    cur = nxt;
  }

  // --- 6. transformer encoder on x_a (h := xa buffers) ---
  float*  h_f32 = xa_f32;
  bf16_t* h_bf  = xa_bf[cur];
  for (int i = 0; i < LENC; ++i) {
    gemm(h_bf, pQkv + (size_t)i * 128 * 384, tr_qkv_b + (size_t)i * 384, nullptr,
         qkv_f32, nullptr, DD, 384, 128, 0);
    attention_kernel<<<8, 64, 0, stream>>>(qkv_f32, o_f32, o_bf);
    gemm(o_bf, pOutW + (size_t)i * 128 * 128, tr_out_b + (size_t)i * 128, h_f32,
         preln, nullptr, DD, 128, 128, 0);
    layernorm_kernel<<<DD, 128, 0, stream>>>(preln, tr_ln1 + (size_t)i * 256, h_f32, h_bf);
    gemm(h_bf, pFf1 + (size_t)i * 128 * FF_, tr_ff1_b + (size_t)i * FF_, nullptr,
         nullptr, f_bf, DD, FF_, 128, 1);
    gemm(f_bf, pFf2 + (size_t)i * FF_ * 128, tr_ff2_b + (size_t)i * 128, h_f32,
         preln, nullptr, DD, 128, FF_, 0);
    layernorm_kernel<<<DD, 128, 0, stream>>>(preln, tr_ln2 + (size_t)i * 256, h_f32, h_bf);
  }

  // --- 7. candidate logits (items read from bf16 state) ---
  hlp_kernel<<<grid1(DD * HH), 256, 0, stream>>>(h_f32, lp_w, hlp);
  logits_kernel<<<grid1((DD - 1) * KC_), 256, 0, stream>>>(xd_bf[cur], hlp, cand_idx, out);
}
